// LSTMEncoder_249108103773
// MI455X (gfx1250) — matrix operand order repaired and checked by reading, not compiled, not run on hardware
//
#include <hip/hip_runtime.h>

// ---------------------------------------------------------------------------
// 2-layer LSTM, T=5, B=4096, H=E=512.
// Fused bf16-WMMA GEMM + LSTM cell per (step, layer); weight strips staged
// in LDS via gfx1250 async-to-LDS copies (ASYNCcnt), fragments via ds_load.
// ---------------------------------------------------------------------------

typedef __attribute__((ext_vector_type(16))) __bf16 v16bf;
typedef __attribute__((ext_vector_type(8)))  float  v8f;

#define HID     512
#define NB      4096
#define NSTEP   5
#define NIN     11
#define KPAD    32
#define MT      2          // 16-row tiles per wave (M direction)
#define WAVES   8
#define THREADS 256
#define LDP     (HID + 8)  // padded LDS row stride (elements): bank-conflict skew

// --- async global -> LDS copy (gfx1250, ASYNCcnt) --------------------------
__device__ __forceinline__ void async_b128(__bf16* lds_dst, const __bf16* gsrc) {
    // Low 32 bits of a flat shared-aperture address are the LDS byte offset.
    unsigned lds_off = (unsigned)(size_t)lds_dst;
    asm volatile("global_load_async_to_lds_b128 %0, %1, off"
                 :: "v"(lds_off), "v"(gsrc) : "memory");
}
__device__ __forceinline__ void wait_async0() {
    asm volatile("s_wait_asynccnt 0" ::: "memory");
}

// --- fragment loaders (layouts per CDNA5 ISA 7.12.2, wave32) ---------------
__device__ __forceinline__ v16bf frag2x16(const __bf16* p0, const __bf16* p1) {
    union { uint4 u[2]; v16bf v; } r;
    r.u[0] = *(const uint4*)p0;
    r.u[1] = *(const uint4*)p1;
    return r.v;
}

// A-matrix 16x32 bf16: lane m = lane&15; halves 0-7 = K kb..kb+7,
// halves 8-15 = K kb+16..kb+23, kb = (lane>=16) ? 8 : 0.  (global path)
__device__ __forceinline__ v16bf load_a(const __bf16* A, int ld, int row0,
                                        int k0, int lane) {
    const int m  = lane & 15;
    const int kb = (lane & 16) ? 8 : 0;
    const __bf16* base = A + (size_t)(row0 + m) * ld + k0 + kb;
    return frag2x16(base, base + 16);
}

// B-matrix 32x16 bf16 from the LDS strip: row (q*16+n) holds W row
// q*HID+n0+n with padded stride ldp; lane n = lane&15 reads K k0+kb..+7
// and k0+kb+16..+23, kb = 8*(lane>=16).
__device__ __forceinline__ v16bf load_b_sh(const __bf16* region, int ldp,
                                           int q, int k0, int lane) {
    const int n  = lane & 15;
    const int kb = (lane & 16) ? 8 : 0;
    const __bf16* p = region + (size_t)(q * 16 + n) * ldp + k0 + kb;
    return frag2x16(p, p + 16);
}

__device__ __forceinline__ float fast_sigmoid(float x) {
    return 1.0f / (1.0f + __expf(-x));
}
__device__ __forceinline__ float fast_tanh(float x) {
    return 2.0f / (1.0f + __expf(-2.0f * x)) - 1.0f;
}

// --- fused gates-GEMM + LSTM cell ------------------------------------------
// gates[B,4H] = Ax @ Wx^T + Ah @ Wh^T + bias; then cell update.
// Ax: [NB, kx*32] bf16, Wx: [4H, kx*32] bf16, Ah/Wh: K=512 (16 chunks).
__global__ __launch_bounds__(THREADS)
void lstm_step_kernel(const __bf16* __restrict__ Ax,
                      const __bf16* __restrict__ Wx, int kx,
                      const __bf16* __restrict__ Ah,
                      const __bf16* __restrict__ Wh,
                      const float*  __restrict__ bias,
                      float*        __restrict__ c,
                      __bf16*       __restrict__ h_out,
                      float*        __restrict__ h_out_f32)
{
    // [X strip: 64 rows x LDP][H strip: 64 rows x LDP]  (130 KB of 320 KB/WGP)
    __shared__ __bf16 ldsW[2 * 64 * LDP];

    const int tid   = threadIdx.x;
    const int lane  = tid & 31;
    const int wave  = tid >> 5;
    const int n_idx = blockIdx.x >> 4;               // 16 blocks per n-strip
    const int m_idx = ((blockIdx.x & 15) << 3) + wave;
    const int m0    = m_idx * (16 * MT);
    const int n0    = n_idx * 16;
    const int ldx   = kx * 32;

    __bf16* shX = ldsW;
    __bf16* shH = ldsW + 64 * LDP;

    // --- cooperative async stage of this block's weight strips -------------
    {
        const int cpr = ldx >> 3;                    // 16B chunks per X row
        for (int i = tid; i < 64 * cpr; i += THREADS) {
            const int row = i / cpr;
            const int j   = (i - row * cpr) << 3;    // element offset in row
            const int q = row >> 4, n = row & 15;
            async_b128(shX + row * LDP + j,
                       Wx + (size_t)(q * HID + n0 + n) * ldx + j);
        }
        for (int i = tid; i < 64 * 64; i += THREADS) {
            const int row = i >> 6;
            const int j   = (i & 63) << 3;
            const int q = row >> 4, n = row & 15;
            async_b128(shH + row * LDP + j,
                       Wh + (size_t)(q * HID + n0 + n) * HID + j);
        }
    }
    wait_async0();
    __syncthreads();

    // accumulators: acc[mi][q], q = gate (i,f,g,o); init with combined bias.
    v8f acc[MT][4];
#pragma unroll
    for (int q = 0; q < 4; ++q) {
        const float bq = bias[q * HID + n0 + (lane & 15)];
#pragma unroll
        for (int mi = 0; mi < MT; ++mi)
#pragma unroll
            for (int e = 0; e < 8; ++e) acc[mi][q][e] = bq;
    }

    // --- x-input GEMM (kx chunks of 32): A global, B from LDS ---
    for (int kc = 0; kc < kx; ++kc) {
        const int k0 = kc * 32;
        const v16bf a0 = load_a(Ax, ldx, m0,      k0, lane);
        const v16bf a1 = load_a(Ax, ldx, m0 + 16, k0, lane);
#pragma unroll
        for (int q = 0; q < 4; ++q) {
            const v16bf bm = load_b_sh(shX, LDP, q, k0, lane);
            acc[0][q] = __builtin_amdgcn_wmma_f32_16x16x32_bf16(
                false, a0, false, bm, (short)0, acc[0][q], false, false);
            acc[1][q] = __builtin_amdgcn_wmma_f32_16x16x32_bf16(
                false, a1, false, bm, (short)0, acc[1][q], false, false);
        }
    }

    // --- recurrent GEMM (K = 512 = 16 chunks): A global, B from LDS ---
    for (int kc = 0; kc < 16; ++kc) {
        const int k0 = kc * 32;
        const v16bf a0 = load_a(Ah, HID, m0,      k0, lane);
        const v16bf a1 = load_a(Ah, HID, m0 + 16, k0, lane);
#pragma unroll
        for (int q = 0; q < 4; ++q) {
            const v16bf bm = load_b_sh(shH, LDP, q, k0, lane);
            acc[0][q] = __builtin_amdgcn_wmma_f32_16x16x32_bf16(
                false, a0, false, bm, (short)0, acc[0][q], false, false);
            acc[1][q] = __builtin_amdgcn_wmma_f32_16x16x32_bf16(
                false, a1, false, bm, (short)0, acc[1][q], false, false);
        }
    }

    // --- LSTM cell update (C layout: VGPR e -> row e + 8*(lane>=16),
    //     col = lane&15) ---
    const int col = n0 + (lane & 15);
#pragma unroll
    for (int mi = 0; mi < MT; ++mi) {
        const int rbase = m0 + 16 * mi + ((lane >> 4) << 3);
#pragma unroll
        for (int e = 0; e < 8; ++e) {
            const size_t idx = (size_t)(rbase + e) * HID + col;
            const float iv = fast_sigmoid(acc[mi][0][e]);
            const float fv = fast_sigmoid(acc[mi][1][e]);
            const float gv = fast_tanh   (acc[mi][2][e]);
            const float ov = fast_sigmoid(acc[mi][3][e]);
            const float cn = fv * c[idx] + iv * gv;
            c[idx] = cn;
            const float hn = ov * fast_tanh(cn);
            h_out[idx] = (__bf16)hn;
            if (h_out_f32) h_out_f32[idx] = hn;
        }
    }
}

// --- prologue kernels -------------------------------------------------------

__global__ void k_cvt_bf16(__bf16* dst, const float* src, int n) {
    const int i = blockIdx.x * blockDim.x + threadIdx.x;
    if (i < n) dst[i] = (__bf16)src[i];
}

// dst[rows][32] <- src[rows][srccols], zero-padded
__global__ void k_pad_rows(__bf16* dst, const float* src, int rows, int srccols) {
    const int i = blockIdx.x * blockDim.x + threadIdx.x;
    if (i < rows * KPAD) {
        const int k = i & (KPAD - 1);
        const int r = i >> 5;
        dst[i] = (k < srccols) ? (__bf16)src[r * srccols + k] : (__bf16)0.0f;
    }
}

// xpad[t][b][32] <- x[b][t][11], zero-padded
__global__ void k_pad_x(__bf16* dst, const float* x) {
    const int i = blockIdx.x * blockDim.x + threadIdx.x;
    if (i < NSTEP * NB * KPAD) {
        const int k = i & (KPAD - 1);
        const int b = (i >> 5) & (NB - 1);
        const int t = i >> 17;                   // NB*KPAD = 2^17
        dst[i] = (k < NIN)
                   ? (__bf16)x[(size_t)b * (NSTEP * NIN) + t * NIN + k]
                   : (__bf16)0.0f;
    }
}

__global__ void k_bias(float* b1, const float* bi1, const float* bh1,
                       float* b2, const float* bi2, const float* bh2) {
    const int i = blockIdx.x * blockDim.x + threadIdx.x;
    if (i < 4 * HID) { b1[i] = bi1[i] + bh1[i]; b2[i] = bi2[i] + bh2[i]; }
}

__global__ void k_zero(uint4* p, int n) {
    const int i = blockIdx.x * blockDim.x + threadIdx.x;
    if (i < n) p[i] = make_uint4(0u, 0u, 0u, 0u);
}

// --- host launcher ----------------------------------------------------------

extern "C" void kernel_launch(void* const* d_in, const int* in_sizes, int n_in,
                              void* d_out, int out_size, void* d_ws, size_t ws_size,
                              hipStream_t stream) {
    const float* x     = (const float*)d_in[0];
    const float* W_ih1 = (const float*)d_in[1];
    const float* W_hh1 = (const float*)d_in[2];
    const float* b_ih1 = (const float*)d_in[3];
    const float* b_hh1 = (const float*)d_in[4];
    const float* W_ih2 = (const float*)d_in[5];
    const float* W_hh2 = (const float*)d_in[6];
    const float* b_ih2 = (const float*)d_in[7];
    const float* b_hh2 = (const float*)d_in[8];

    char* ws = (char*)d_ws;
    size_t off = 0;
    auto take = [&](size_t bytes) -> char* {
        char* p = ws + off;
        off = (off + bytes + 255) & ~(size_t)255;
        return p;
    };

    const size_t WBYTES = (size_t)4 * HID * HID * 2;   // 2 MB bf16 weight
    const size_t HBYTES = (size_t)NB * HID * 2;        // 4 MB bf16 state
    const size_t CBYTES = (size_t)NB * HID * 4;        // 8 MB f32 state

    __bf16* whh1  = (__bf16*)take(WBYTES);
    __bf16* wih2  = (__bf16*)take(WBYTES);
    __bf16* whh2  = (__bf16*)take(WBYTES);
    __bf16* wih1p = (__bf16*)take((size_t)4 * HID * KPAD * 2);
    __bf16* xpad  = (__bf16*)take((size_t)NSTEP * NB * KPAD * 2);
    float*  bias1 = (float*)take((size_t)4 * HID * 4);
    float*  bias2 = (float*)take((size_t)4 * HID * 4);
    // zero region: h0[0], h1[0], c0, c1 contiguous (sizes are multiples of
    // 256, so take() leaves no gaps).
    __bf16* h0buf0 = (__bf16*)take(HBYTES);
    __bf16* h1buf0 = (__bf16*)take(HBYTES);
    float*  c0     = (float*)take(CBYTES);
    float*  c1     = (float*)take(CBYTES);
    __bf16* h0buf1 = (__bf16*)take(HBYTES);
    __bf16* h1buf1 = (__bf16*)take(HBYTES);
    (void)in_sizes; (void)n_in; (void)out_size; (void)ws_size;

    __bf16* h0buf[2] = { h0buf0, h0buf1 };
    __bf16* h1buf[2] = { h1buf0, h1buf1 };

    const int nW = 4 * HID * HID;                       // 1,048,576
    k_cvt_bf16<<<(nW + 255) / 256, 256, 0, stream>>>(whh1, W_hh1, nW);
    k_cvt_bf16<<<(nW + 255) / 256, 256, 0, stream>>>(wih2, W_ih2, nW);
    k_cvt_bf16<<<(nW + 255) / 256, 256, 0, stream>>>(whh2, W_hh2, nW);
    k_pad_rows<<<(4 * HID * KPAD + 255) / 256, 256, 0, stream>>>(
        wih1p, W_ih1, 4 * HID, NIN);
    k_pad_x<<<(NSTEP * NB * KPAD + 255) / 256, 256, 0, stream>>>(xpad, x);
    k_bias<<<(4 * HID + 255) / 256, 256, 0, stream>>>(
        bias1, b_ih1, b_hh1, bias2, b_ih2, b_hh2);
    const int nz = (int)((2 * HBYTES + 2 * CBYTES) / 16);
    k_zero<<<(nz + 255) / 256, 256, 0, stream>>>((uint4*)h0buf0, nz);

    const int NBLK = (NB / (16 * MT)) * (HID / 16) / WAVES;  // 512

    for (int t = 0; t < NSTEP; ++t) {
        const __bf16* h0r = h0buf[t & 1];
        __bf16*       h0w = h0buf[(t + 1) & 1];
        const __bf16* h1r = h1buf[t & 1];
        __bf16*       h1w = h1buf[(t + 1) & 1];

        // layer 1: x-part is the padded K=32 chunk
        lstm_step_kernel<<<NBLK, THREADS, 0, stream>>>(
            xpad + (size_t)t * NB * KPAD, wih1p, 1,
            h0r, whh1, bias1, c0, h0w, nullptr);

        // layer 2: x-input is h0 (just written), K=512
        lstm_step_kernel<<<NBLK, THREADS, 0, stream>>>(
            h0w, wih2, 16,
            h1r, whh2, bias2, c1, h1w,
            (t == NSTEP - 1) ? (float*)d_out : nullptr);
    }
}
